// Glm4MoeMLPStack_3246995276390
// MI455X (gfx1250) — matrix operand order repaired and checked by reading, not compiled, not run on hardware
//
#include <hip/hip_runtime.h>
#include <hip/hip_bf16.h>

// MoE expert-MLP for MI455X (gfx1250, wave32, WMMA).
//   1) hidden = silu(x@gate_w) * (x@up_w)  -> bf16 in d_ws  (T*I*2 = 96 MB)
//   2) out    = hidden @ down_w            -> fp32 d_out
// fp32 -> bf16 conversion during staging; v_wmma_f32_16x16x32_bf16 w/ f32 acc.
// Block tile 128x128, BK=32, 8 wave32s (2x4); wave tile 64x32 = 4x2 fragments.
// Software pipeline: next K-tile global->regs (packed bf16) overlapped with
// WMMA on current LDS tile; ping-pong LDS buffers -> one barrier per K-step.

constexpr int kE = 16;
constexpr int kH = 2048;
constexpr int kI = 1536;
constexpr int kT = 32768;

constexpr int BM  = 128;
constexpr int BN  = 128;
constexpr int BK  = 32;   // K per iteration == WMMA K
constexpr int BKP = 40;   // padded LDS row stride (80 B: 16B-aligned, bank-spread)

typedef __attribute__((ext_vector_type(16))) __bf16        v16bf;
typedef __attribute__((ext_vector_type(8)))  float         v8f;
typedef __attribute__((ext_vector_type(8)))  unsigned int  v8u;

static __device__ __forceinline__ unsigned pack2(float a, float b) {
  __bf16 x = (__bf16)a, y = (__bf16)b;
  unsigned short ux = __builtin_bit_cast(unsigned short, x);
  unsigned short uy = __builtin_bit_cast(unsigned short, y);
  return (unsigned)ux | ((unsigned)uy << 16);
}

// A fragment (16x32 bf16, MxK): lanes 0-15 = row M=lane, K 0..7 / 16..23;
// lanes 16-31 = K 8..15 / 24..31. LDS row-major, stride BKP bf16.
static __device__ __forceinline__ v16bf ld_frag_a(const __bf16* tile, int rowOff, int lane) {
  int r = rowOff + (lane & 15);
  int h = lane >> 4;
  const __bf16* p = tile + r * BKP + h * 8;
  uint4 lo = *(const uint4*)(p);
  uint4 hi = *(const uint4*)(p + 16);
  v8u u;
  u[0] = lo.x; u[1] = lo.y; u[2] = lo.z; u[3] = lo.w;
  u[4] = hi.x; u[5] = hi.y; u[6] = hi.z; u[7] = hi.w;
  return __builtin_bit_cast(v16bf, u);
}

// B fragment (32x16 bf16, KxN): lanes 0-15 = col N=lane, K 0..7 / 16..23; lanes 16-31
// = K 8..15 / 24..31. LDS stored N-major (B^T rows, K contiguous), stride BKP bf16.
static __device__ __forceinline__ v16bf ld_frag_b(const __bf16* tile, int colOff, int lane) {
  int n = colOff + (lane & 15);
  int h = lane >> 4;
  const __bf16* p = tile + n * BKP + h * 8;
  uint4 lo = *(const uint4*)(p);
  uint4 hi = *(const uint4*)(p + 16);
  v8u u;
  u[0] = lo.x; u[1] = lo.y; u[2] = lo.z; u[3] = lo.w;
  u[4] = hi.x; u[5] = hi.y; u[6] = hi.z; u[7] = hi.w;
  return __builtin_bit_cast(v16bf, u);
}

static __device__ __forceinline__ int expert_of_row(const int* __restrict__ gs, int row0) {
  int e = 0;
  int csum = gs[0];
  while (e < kE - 1 && row0 >= csum) { ++e; csum += gs[e]; }
  return e;
}

// Stage a 16-float (half K-row) A strip: fp32 global -> 8 packed bf16 dwords.
static __device__ __forceinline__ void load_a_f32(const float* p, unsigned a[8]) {
  const float4* q = (const float4*)p;
  float4 f0 = q[0], f1 = q[1], f2 = q[2], f3 = q[3];
  a[0] = pack2(f0.x, f0.y); a[1] = pack2(f0.z, f0.w);
  a[2] = pack2(f1.x, f1.y); a[3] = pack2(f1.z, f1.w);
  a[4] = pack2(f2.x, f2.y); a[5] = pack2(f2.z, f2.w);
  a[6] = pack2(f3.x, f3.y); a[7] = pack2(f3.z, f3.w);
}

// Stage a (k,k+1) x 8-col strip of a weight matrix: two coalesced float4 pairs
// from rows k and k+1, packed so dword j = cols nb+j holding {k, k+1}.
static __device__ __forceinline__ void load_b_f32(const float* w, size_t base, int ldw,
                                                  unsigned b[8]) {
  const float4* r0 = (const float4*)(w + base);
  const float4* r1 = (const float4*)(w + base + ldw);
  float4 a0 = r0[0], a1 = r0[1], b0 = r1[0], b1 = r1[1];
  b[0] = pack2(a0.x, b0.x); b[1] = pack2(a0.y, b0.y);
  b[2] = pack2(a0.z, b0.z); b[3] = pack2(a0.w, b0.w);
  b[4] = pack2(a1.x, b1.x); b[5] = pack2(a1.y, b1.y);
  b[6] = pack2(a1.z, b1.z); b[7] = pack2(a1.w, b1.w);
}

static __device__ __forceinline__ void store_b_lds(__bf16* tile, int nb, int kp,
                                                   const unsigned b[8]) {
  unsigned* d = (unsigned*)tile;
#pragma unroll
  for (int j = 0; j < 8; ++j) d[(nb + j) * (BKP / 2) + kp] = b[j];
}

// ---------------------------------------------------------------------------
// Kernel 1: hidden = silu(x@gate_w) * (x@up_w), bf16 output.
// grid = (kI/BN, kT/BM), block = 256.
// ---------------------------------------------------------------------------
__global__ __launch_bounds__(256) void moe_gateup_silu(
    const float* __restrict__ x,
    const float* __restrict__ gate_w,
    const float* __restrict__ up_w,
    const int*   __restrict__ gs,
    unsigned short* __restrict__ hidden_raw) {
  __shared__ __align__(16) __bf16 As[2][BM * BKP];
  __shared__ __align__(16) __bf16 Bg[2][BN * BKP];
  __shared__ __align__(16) __bf16 Bu[2][BN * BKP];
  __bf16* hidden = reinterpret_cast<__bf16*>(hidden_raw);

  const int tid  = threadIdx.x;
  const int lane = tid & 31;
  const int wave = tid >> 5;
  const int wm   = wave >> 2;  // 0..1  -> M offset wm*64
  const int wn   = wave & 3;   // 0..3  -> N offset wn*32

  const int row0 = blockIdx.y * BM;
  const int n0   = blockIdx.x * BN;

  const int e = expert_of_row(gs, row0);
  const float* wg = gate_w + (size_t)e * kH * kI;
  const float* wu = up_w   + (size_t)e * kH * kI;

  const v8f vz = {0.f, 0.f, 0.f, 0.f, 0.f, 0.f, 0.f, 0.f};
  v8f accg[4][2], accu[4][2];
#pragma unroll
  for (int i = 0; i < 4; ++i) {
    accg[i][0] = vz; accg[i][1] = vz;
    accu[i][0] = vz; accu[i][1] = vz;
  }

  const int am = tid >> 1;          // A: this thread's row
  const int ah = tid & 1;           // A: which 16-float half of the K-row
  const float* xa = x + (size_t)(row0 + am) * kH + ah * 16;
  const int kp = tid & 15;          // B: k-pair (k = 2kp, 2kp+1)
  const int nb = (tid >> 4) * 8;    // B: 8-column strip

  constexpr int NIT = kH / BK;      // 64
  unsigned sa[8], sg[8], su[8];

  // Prologue: stage K-tile 0 into registers.
  load_a_f32(xa, sa);
  {
    size_t base = (size_t)(2 * kp) * kI + n0 + nb;
    load_b_f32(wg, base, kI, sg);
    load_b_f32(wu, base, kI, su);
  }

  int buf = 0;
  for (int it = 0; it < NIT; ++it) {
    // ---- commit staged regs to LDS[buf] ----
    {
      unsigned* dst = (unsigned*)(As[buf] + am * BKP) + ah * 8;
#pragma unroll
      for (int j = 0; j < 8; ++j) dst[j] = sa[j];
      store_b_lds(Bg[buf], nb, kp, sg);
      store_b_lds(Bu[buf], nb, kp, su);
    }
    __syncthreads();

    // ---- issue next tile's global loads (overlap with WMMA below) ----
    if (it + 1 < NIT) {
      int k1 = (it + 1) * BK;
      load_a_f32(xa + k1, sa);
      size_t base = (size_t)(k1 + 2 * kp) * kI + n0 + nb;
      load_b_f32(wg, base, kI, sg);
      load_b_f32(wu, base, kI, su);
      if (it + 2 < NIT) {  // distance-2 L2 prefetch of the streaming operands
        int k2 = (it + 2) * BK;
        size_t pb = (size_t)(k2 + 2 * kp) * kI + n0 + nb;
        __builtin_prefetch(xa + k2, 0, 1);
        __builtin_prefetch(wg + pb, 0, 1);
        __builtin_prefetch(wu + pb, 0, 1);
      }
    }

    // ---- compute on LDS[buf] ----
    v16bf afr[4];
#pragma unroll
    for (int fm = 0; fm < 4; ++fm)
      afr[fm] = ld_frag_a(As[buf], wm * 64 + fm * 16, lane);
    v16bf bgfr[2], bufr[2];
#pragma unroll
    for (int fn = 0; fn < 2; ++fn) {
      bgfr[fn] = ld_frag_b(Bg[buf], wn * 32 + fn * 16, lane);
      bufr[fn] = ld_frag_b(Bu[buf], wn * 32 + fn * 16, lane);
    }
#pragma unroll
    for (int fm = 0; fm < 4; ++fm) {
#pragma unroll
      for (int fn = 0; fn < 2; ++fn) {
        accg[fm][fn] = __builtin_amdgcn_wmma_f32_16x16x32_bf16(
            false, afr[fm], false, bgfr[fn], (short)0, accg[fm][fn], false, false);
        accu[fm][fn] = __builtin_amdgcn_wmma_f32_16x16x32_bf16(
            false, afr[fm], false, bufr[fn], (short)0, accu[fm][fn], false, false);
      }
    }
    buf ^= 1;
  }

  // Epilogue: silu(g)*u -> bf16. Fast sigmoid via v_rcp_f32 (no IEEE div).
  const int hh = lane >> 4;
  const int nn = lane & 15;
#pragma unroll
  for (int fm = 0; fm < 4; ++fm) {
#pragma unroll
    for (int fn = 0; fn < 2; ++fn) {
#pragma unroll
      for (int i = 0; i < 8; ++i) {
        float g = accg[fm][fn][i];
        float u = accu[fm][fn][i];
        float s = g * __builtin_amdgcn_rcpf(1.0f + __expf(-g));
        float hv = s * u;
        int row = row0 + wm * 64 + fm * 16 + hh * 8 + i;
        int col = n0 + wn * 32 + fn * 16 + nn;
        hidden[(size_t)row * kI + col] = (__bf16)hv;
      }
    }
  }
}

// ---------------------------------------------------------------------------
// Kernel 2: out = hidden @ down_w, fp32 output.
// grid = (kH/BN, kT/BM), block = 256.
// ---------------------------------------------------------------------------
__global__ __launch_bounds__(256) void moe_down(
    const unsigned short* __restrict__ hidden_raw,
    const float* __restrict__ down_w,
    const int*   __restrict__ gs,
    float* __restrict__ out) {
  __shared__ __align__(16) __bf16 As[2][BM * BKP];
  __shared__ __align__(16) __bf16 Bs[2][BN * BKP];
  const __bf16* hidden = reinterpret_cast<const __bf16*>(hidden_raw);

  const int tid  = threadIdx.x;
  const int lane = tid & 31;
  const int wave = tid >> 5;
  const int wm   = wave >> 2;
  const int wn   = wave & 3;

  const int row0 = blockIdx.y * BM;
  const int n0   = blockIdx.x * BN;

  const int e = expert_of_row(gs, row0);
  const float* wd = down_w + (size_t)e * kI * kH;

  const v8f vz = {0.f, 0.f, 0.f, 0.f, 0.f, 0.f, 0.f, 0.f};
  v8f acc[4][2];
#pragma unroll
  for (int i = 0; i < 4; ++i) { acc[i][0] = vz; acc[i][1] = vz; }

  const int am = tid >> 1;
  const int ah = tid & 1;
  const __bf16* ha = hidden + (size_t)(row0 + am) * kI;
  const int kp = tid & 15;
  const int nb = (tid >> 4) * 8;

  constexpr int NIT = kI / BK;      // 48
  uint4 qa0, qa1;                   // A is already bf16: raw 32B strip
  unsigned sb[8];

  {
    const uint4* p = (const uint4*)(ha) + ah * 2;
    qa0 = p[0]; qa1 = p[1];
    load_b_f32(wd, (size_t)(2 * kp) * kH + n0 + nb, kH, sb);
  }

  int buf = 0;
  for (int it = 0; it < NIT; ++it) {
    {
      uint4* da = (uint4*)(As[buf] + am * BKP) + ah * 2;  // 80B rows, 16B aligned
      da[0] = qa0; da[1] = qa1;
      store_b_lds(Bs[buf], nb, kp, sb);
    }
    __syncthreads();

    if (it + 1 < NIT) {
      int k1 = (it + 1) * BK;
      const uint4* p = (const uint4*)(ha + k1) + ah * 2;
      qa0 = p[0]; qa1 = p[1];
      load_b_f32(wd, (size_t)(k1 + 2 * kp) * kH + n0 + nb, kH, sb);
      if (it + 2 < NIT) {
        int k2 = (it + 2) * BK;
        __builtin_prefetch(ha + k2, 0, 1);
        __builtin_prefetch(wd + (size_t)(k2 + 2 * kp) * kH + n0 + nb, 0, 1);
      }
    }

    v16bf afr[4];
#pragma unroll
    for (int fm = 0; fm < 4; ++fm)
      afr[fm] = ld_frag_a(As[buf], wm * 64 + fm * 16, lane);
    v16bf bfr[2];
#pragma unroll
    for (int fn = 0; fn < 2; ++fn)
      bfr[fn] = ld_frag_b(Bs[buf], wn * 32 + fn * 16, lane);
#pragma unroll
    for (int fm = 0; fm < 4; ++fm) {
#pragma unroll
      for (int fn = 0; fn < 2; ++fn) {
        acc[fm][fn] = __builtin_amdgcn_wmma_f32_16x16x32_bf16(
            false, afr[fm], false, bfr[fn], (short)0, acc[fm][fn], false, false);
      }
    }
    buf ^= 1;
  }

  const int hh = lane >> 4;
  const int nn = lane & 15;
#pragma unroll
  for (int fm = 0; fm < 4; ++fm) {
#pragma unroll
    for (int fn = 0; fn < 2; ++fn) {
#pragma unroll
      for (int i = 0; i < 8; ++i) {
        int row = row0 + wm * 64 + fm * 16 + hh * 8 + i;
        int col = n0 + wn * 32 + fn * 16 + nn;
        out[(size_t)row * kH + col] = acc[fm][fn][i];
      }
    }
  }
}

extern "C" void kernel_launch(void* const* d_in, const int* in_sizes, int n_in,
                              void* d_out, int out_size, void* d_ws, size_t ws_size,
                              hipStream_t stream) {
  const float* x      = (const float*)d_in[0];
  const float* gate_w = (const float*)d_in[1];
  const float* up_w   = (const float*)d_in[2];
  const float* down_w = (const float*)d_in[3];
  const int*   gs     = (const int*)d_in[4];
  float* out = (float*)d_out;
  // Scratch: hidden activations, bf16, T*I = 32768*1536 -> 96 MB.
  unsigned short* hidden = (unsigned short*)d_ws;

  dim3 g1(kI / BN, kT / BM);  // 12 x 256
  moe_gateup_silu<<<g1, 256, 0, stream>>>(x, gate_w, up_w, gs, hidden);

  dim3 g2(kH / BN, kT / BM);  // 16 x 256
  moe_down<<<g2, 256, 0, stream>>>(hidden, down_w, gs, out);
}
